// AttentionBlock_14955076125230
// MI455X (gfx1250) — matrix operand order repaired and checked by reading, not compiled, not run on hardware
//
#include <hip/hip_runtime.h>
#include <stdint.h>

// ---------------------------------------------------------------- constants
#define B_N   4
#define C_N   512
#define T_N   2048
#define G_N   8
#define H_N   8
#define CH_N  64
#define O3_N  1536          // 3*C
#define EPS_F 1e-5f

// ---------------------------------------------------------------- WMMA types
typedef __bf16 bf16_t;
typedef bf16_t bf16x16 __attribute__((ext_vector_type(16)));
typedef float  floatx8 __attribute__((ext_vector_type(8)));

union FragAB {
    bf16x16  v;
    uint16_t e[16];
    uint4    q[2];
};

__device__ __forceinline__ uint16_t f2bf(float f) {
    union { float f; uint32_t u; } x; x.f = f;
    uint32_t r = x.u + 0x7FFFu + ((x.u >> 16) & 1u);   // round-to-nearest-even
    return (uint16_t)(r >> 16);
}
__device__ __forceinline__ float bf2f(uint16_t h) {
    union { float f; uint32_t u; } x; x.u = (uint32_t)h << 16;
    return x.f;
}
__device__ __forceinline__ floatx8 zero8() {
    floatx8 z;
#pragma unroll
    for (int i = 0; i < 8; ++i) z[i] = 0.f;
    return z;
}

// ------------------------------------------------ CDNA5 async global->LDS copy
// Generic pointers to LDS carry the LDS offset in their low 32 bits.
__device__ __forceinline__ uint32_t lds_off(const void* p) {
    return (uint32_t)(uintptr_t)p;
}
__device__ __forceinline__ void async_load_lds_b128(uint32_t dst_lds, const void* src) {
    asm volatile("global_load_async_to_lds_b128 %0, %1, off"
                 :: "v"(dst_lds), "v"(src) : "memory");
}
__device__ __forceinline__ void wait_async0() {
#if __has_builtin(__builtin_amdgcn_s_wait_asynccnt)
    __builtin_amdgcn_s_wait_asynccnt(0);
#else
    asm volatile("s_wait_asynccnt 0x0" ::: "memory");
#endif
}

// ------------------------------------------------ CDNA5 Tensor Data Mover (2D)
#if __has_builtin(__builtin_amdgcn_tensor_load_to_lds)
#define HAVE_TDM 1
typedef unsigned int uint32x4 __attribute__((ext_vector_type(4)));
typedef int          int32x4  __attribute__((ext_vector_type(4)));
typedef int          int32x8  __attribute__((ext_vector_type(8)));

// 2D tile load: tile_d1 rows of tile_d0 elements (2B each), row stride in elems.
// Descriptor bitfields per CDNA5 ISA.
// This toolchain (clang-23 / therock-10.0 headers) uses the 6-arg builtin form.
__device__ __forceinline__ void tdm_load_2d(uint32_t dst_lds, const void* gptr,
                                            uint32_t tile_d0, uint32_t tile_d1,
                                            uint32_t row_stride) {
    uint64_t ga = (uint64_t)(uintptr_t)gptr;
    uint32x4 g0;
    g0[0] = 1u;                                            // count=1 (valid user D#)
    g0[1] = dst_lds;                                       // lds_addr   [63:32]
    g0[2] = (uint32_t)ga;                                  // global_addr[31:0]
    g0[3] = (uint32_t)((ga >> 32) & 0x1FFFFFFu) | (2u << 30); // addr[56:32] | type=2
    int32x8 g1;
    g1[0] = (int)(1u << 16);                               // data_size=1 (2 bytes)
    g1[1] = (int)((tile_d0 & 0xFFFFu) << 16);              // tensor_dim0[15:0]
    g1[2] = (int)((tile_d1 & 0xFFFFu) << 16);              // dim0 hi | tensor_dim1 lo
    g1[3] = (int)(tile_d0 << 16);                          // dim1 hi | tile_dim0
    g1[4] = (int)(tile_d1 & 0xFFFFu);                      // tile_dim1 | tile_dim2=0
    g1[5] = (int)row_stride;                               // tensor_dim0_stride[31:0]
    g1[6] = 0;                                             // stride hi | dim1_stride lo
    g1[7] = 0;
    int32x4 z4; z4[0] = z4[1] = z4[2] = z4[3] = 0;
    int32x8 z8;
#pragma unroll
    for (int i = 0; i < 8; ++i) z8[i] = 0;
    __builtin_amdgcn_tensor_load_to_lds(g0, g1, z4, z4, z8, 0);
}
__device__ __forceinline__ void wait_tensor0() {
    __builtin_amdgcn_s_wait_tensorcnt(0);
}
#else
#define HAVE_TDM 0
#endif

// ---------------------------------------------------------------- GN stats
__global__ __launch_bounds__(256) void gn_stats_kernel(const float* __restrict__ x,
                                                       float* __restrict__ stats) {
    const int blk = blockIdx.x;                        // b*G + g
    const size_t n = (size_t)(C_N / G_N) * T_N;        // 131072 contiguous floats
    const float* p = x + (size_t)blk * n;
    float s = 0.f, ss = 0.f;
    for (size_t i = threadIdx.x; i < n; i += 256) {
        float v = p[i]; s += v; ss += v * v;
    }
    __shared__ float rs[256], rq[256];
    rs[threadIdx.x] = s; rq[threadIdx.x] = ss;
    __syncthreads();
    for (int off = 128; off > 0; off >>= 1) {
        if ((int)threadIdx.x < off) {
            rs[threadIdx.x] += rs[threadIdx.x + off];
            rq[threadIdx.x] += rq[threadIdx.x + off];
        }
        __syncthreads();
    }
    if (threadIdx.x == 0) {
        float mean = rs[0] / (float)n;
        float var  = rq[0] / (float)n - mean * mean;
        stats[blk]      = mean;
        stats[32 + blk] = rsqrtf(var + EPS_F);
    }
}

// ---------------------------------------------------------------- GN apply -> bf16 h
__global__ __launch_bounds__(256) void gn_apply_kernel(const float* __restrict__ x,
                                                       const float* __restrict__ gw,
                                                       const float* __restrict__ gb,
                                                       const float* __restrict__ stats,
                                                       uint16_t* __restrict__ hbf) {
    size_t i = (size_t)blockIdx.x * 256 + threadIdx.x;      // over B*C*T
    int c = (int)((i / T_N) % C_N);
    int b = (int)(i / ((size_t)C_N * T_N));
    int g = c / (C_N / G_N);
    float mean = stats[b * G_N + g];
    float rstd = stats[32 + b * G_N + g];
    float h = (x[i] - mean) * rstd * gw[c] + gb[c];
    hbf[i] = f2bf(h);
}

// ---------------------------------------------------------------- f32 -> bf16 weights
__global__ __launch_bounds__(256) void cvt_bf_kernel(const float* __restrict__ src,
                                                     uint16_t* __restrict__ dst, int n) {
    int i = blockIdx.x * 256 + threadIdx.x;
    if (i < n) dst[i] = f2bf(src[i]);
}

// ---------------------------------------------------------------- tiled bf16 GEMM
// Out[b][o][t] = sum_c W[o][c] * In[b][c][t] + bias[o]   (MODE 0: bf16 out)
//                                                         (MODE 1: f32 out = resid + val)
// block = 256 thr (8 waves), 128(O) x 128(T) tile, K stepped by 32.
// W tile staged by the Tensor Data Mover (fallback: async global->LDS b128).
template <int MODE>
__global__ __launch_bounds__(256) void gemm_kernel(const uint16_t* __restrict__ Wbf,
                                                   const uint16_t* __restrict__ Inbf,
                                                   const float* __restrict__ bias,
                                                   uint16_t* __restrict__ outbf,
                                                   const float* __restrict__ resid,
                                                   float* __restrict__ outf,
                                                   int Kdim, int Odim) {
    __shared__ uint16_t Wt [128 * 32];   // [o_local][c_local]
    __shared__ uint16_t InT[128 * 32];   // [t_local][c_local]  (transposed)

    const int tid    = threadIdx.x;
    const int b      = blockIdx.z;
    const int tblock = blockIdx.x * 128;
    const int oblock = blockIdx.y * 128;

    const int wave = tid >> 5, lane = tid & 31;
    const int wm = wave >> 1, wn = wave & 1;         // 4x2 wave grid
    const int lm = lane & 15, lh = lane >> 4;

    floatx8 acc[2][4];
#pragma unroll
    for (int i = 0; i < 2; ++i)
#pragma unroll
        for (int j = 0; j < 4; ++j) acc[i][j] = zero8();

    for (int kb = 0; kb < Kdim; kb += 32) {
        __syncthreads();
#if HAVE_TDM
        if (wave == 0) {   // one TDM issue per block: 128 rows x 64B, stride Kdim*2B
            tdm_load_2d(lds_off(Wt), Wbf + (size_t)oblock * Kdim + kb,
                        32u, 128u, (uint32_t)Kdim);
        }
#else
        {   // async stage of W tile 128x32 (two b128 per thread)
            int oo = tid >> 1;
            int c0 = (tid & 1) * 16;
            const uint16_t* src = Wbf + (size_t)(oblock + oo) * Kdim + kb + c0;
            async_load_lds_b128(lds_off(&Wt[oo * 32 + c0]), src);
            async_load_lds_b128(lds_off(&Wt[oo * 32 + c0 + 8]), src + 8);
        }
#endif
        {   // stage In tile 32x128 -> transposed [t][c]
            int cc = tid >> 3;              // 0..31
            int t0 = (tid & 7) * 16;        // 0..112
            const uint16_t* src = Inbf + ((size_t)b * Kdim + kb + cc) * T_N + tblock + t0;
            if (kb + 32 < Kdim)             // global_prefetch next K tile
                __builtin_prefetch(src + (size_t)32 * T_N, 0, 1);
#pragma unroll
            for (int j = 0; j < 16; ++j) InT[(t0 + j) * 32 + cc] = src[j];
        }
#if HAVE_TDM
        if (wave == 0) wait_tensor0();
#else
        wait_async0();
#endif
        __syncthreads();

        FragAB afr[2];
#pragma unroll
        for (int mi = 0; mi < 2; ++mi) {     // A: lane=row, K 0-7/16-23 | 8-15/24-31
            int row = wm * 32 + mi * 16 + lm;
            const uint16_t* p = &Wt[row * 32 + lh * 8];
            afr[mi].q[0] = *(const uint4*)p;
            afr[mi].q[1] = *(const uint4*)(p + 16);
        }
#pragma unroll
        for (int ni = 0; ni < 4; ++ni) {     // B: lane=col, K 0-7/16-23 | 8-15/24-31
            FragAB bfr;
            int tcol = wn * 64 + ni * 16 + lm;
            const uint16_t* p = &InT[tcol * 32 + lh * 8];
            bfr.q[0] = *(const uint4*)p;
            bfr.q[1] = *(const uint4*)(p + 16);
#pragma unroll
            for (int mi = 0; mi < 2; ++mi)
                acc[mi][ni] = __builtin_amdgcn_wmma_f32_16x16x32_bf16(
                    false, afr[mi].v, false, bfr.v, (short)0, acc[mi][ni], false, false);
        }
    }

    const int mbase = lh * 8;                // C/D: m = vgpr + 8*lane_half, n = lane&15
#pragma unroll
    for (int mi = 0; mi < 2; ++mi) {
#pragma unroll
        for (int r = 0; r < 8; ++r) {
            int o = oblock + wm * 32 + mi * 16 + mbase + r;
            float bv = bias[o];
#pragma unroll
            for (int ni = 0; ni < 4; ++ni) {
                int t = tblock + wn * 64 + ni * 16 + lm;
                float val = acc[mi][ni][r] + bv;
                size_t idx = ((size_t)b * Odim + o) * T_N + t;
                if (MODE == 0) outbf[idx] = f2bf(val);
                else           outf[idx]  = resid[idx] + val;
            }
        }
    }
}

// ---------------------------------------------------------------- flash attention
// One block = one head-batch x 128 queries; 8 waves x 16 queries each.
// S = (q/8)^T k via WMMA, online softmax, A = P V^T via WMMA.
// V tile staged with async global->LDS b128 (no transpose needed).
__global__ __launch_bounds__(256) void attn_kernel(const uint16_t* __restrict__ qkv,
                                                   uint16_t* __restrict__ abf) {
    __shared__ uint16_t qT[128 * 64];        // [t_local][c]  (q transposed, pre-scaled)
    __shared__ uint16_t kT[32 * 64];         // [s_local][c]  (k transposed)
    __shared__ uint16_t vT[64 * 32];         // [c][s_local]
    __shared__ uint16_t pbuf[8 * 16 * 32];   // per-wave P tile [m][s_local]

    const int tid    = threadIdx.x;
    const int bh     = blockIdx.y;
    const int b      = bh >> 3, hd = bh & 7;
    const int tblock = blockIdx.x * 128;

    const uint16_t* qp = qkv + ((size_t)b * O3_N + hd * (3 * CH_N)) * T_N;
    const uint16_t* kp = qp + (size_t)CH_N * T_N;
    const uint16_t* vp = qp + (size_t)(2 * CH_N) * T_N;

    {   // stage qT with exact 1/8 scale (scale^2 of the reference) folded in
        int c  = tid >> 2;                  // 0..63
        int t0 = (tid & 3) * 32;            // 0,32,64,96
        const uint16_t* src = qp + (size_t)c * T_N + tblock + t0;
#pragma unroll
        for (int j = 0; j < 32; ++j)
            qT[(t0 + j) * 64 + c] = f2bf(bf2f(src[j]) * 0.125f);
    }
    __syncthreads();

    const int wave = tid >> 5, lane = tid & 31;
    const int lm = lane & 15, lh = lane >> 4;

    FragAB aq[2];                            // q fragments, fixed for whole loop
#pragma unroll
    for (int h = 0; h < 2; ++h) {
        int trow = wave * 16 + lm;
        const uint16_t* p = &qT[trow * 64 + h * 32 + lh * 8];
        aq[h].q[0] = *(const uint4*)p;
        aq[h].q[1] = *(const uint4*)(p + 16);
    }

    floatx8 acc2[4];
#pragma unroll
    for (int ni = 0; ni < 4; ++ni) acc2[ni] = zero8();
    float m_run[8], l_run[8];
#pragma unroll
    for (int r = 0; r < 8; ++r) { m_run[r] = -3.0e38f; l_run[r] = 0.f; }

    for (int s = 0; s < T_N; s += 32) {
        __syncthreads();
        {   // k tile: scalar transpose into LDS; v tile: async b128 copy
            int c  = tid >> 2;              // 0..63
            int s0 = (tid & 3) * 8;         // 0,8,16,24
            const uint16_t* ks = kp + (size_t)c * T_N + s + s0;
#pragma unroll
            for (int j = 0; j < 8; ++j) kT[(s0 + j) * 64 + c] = ks[j];
            const uint16_t* vs = vp + (size_t)c * T_N + s + s0;
            async_load_lds_b128(lds_off(&vT[c * 32 + s0]), vs);
        }
        wait_async0();
        __syncthreads();

        // ---- scores: two 16x16 D tiles, K=64 contraction in two WMMAs each
        floatx8 sc[2];
#pragma unroll
        for (int sn = 0; sn < 2; ++sn) {
            floatx8 z = zero8();
#pragma unroll
            for (int h = 0; h < 2; ++h) {
                FragAB bk;
                int scol = sn * 16 + lm;
                const uint16_t* p = &kT[scol * 64 + h * 32 + lh * 8];
                bk.q[0] = *(const uint4*)p;
                bk.q[1] = *(const uint4*)(p + 16);
                z = __builtin_amdgcn_wmma_f32_16x16x32_bf16(
                        false, aq[h].v, false, bk.v, (short)0, z, false, false);
            }
            sc[sn] = z;
        }

        // ---- online softmax (row m = lh*8 + r lives in one 16-lane half)
#pragma unroll
        for (int r = 0; r < 8; ++r) {
            float mx = fmaxf(sc[0][r], sc[1][r]);
#pragma unroll
            for (int off = 1; off <= 8; off <<= 1)
                mx = fmaxf(mx, __shfl_xor(mx, off, 32));
            float nm    = fmaxf(m_run[r], mx);
            float alpha = __expf(m_run[r] - nm);
            float p0 = __expf(sc[0][r] - nm);
            float p1 = __expf(sc[1][r] - nm);
            sc[0][r] = p0; sc[1][r] = p1;
            float rs = p0 + p1;
#pragma unroll
            for (int off = 1; off <= 8; off <<= 1)
                rs += __shfl_xor(rs, off, 32);
            l_run[r] = l_run[r] * alpha + rs;
            m_run[r] = nm;
#pragma unroll
            for (int ni = 0; ni < 4; ++ni) acc2[ni][r] *= alpha;
        }

        // ---- P: C/D layout -> A layout via wave-private LDS round trip
        uint16_t* pw = &pbuf[wave * 16 * 32];
#pragma unroll
        for (int r = 0; r < 8; ++r) {
            int m = lh * 8 + r;
            pw[m * 32 + lm]      = f2bf(sc[0][r]);
            pw[m * 32 + 16 + lm] = f2bf(sc[1][r]);
        }
        FragAB ap;
        {
            const uint16_t* p = &pw[lm * 32 + lh * 8];
            ap.q[0] = *(const uint4*)p;
            ap.q[1] = *(const uint4*)(p + 16);
        }

        // ---- out^T[t][c] += P (16x32) x v^T (32x64): 4 N-tiles
#pragma unroll
        for (int ni = 0; ni < 4; ++ni) {
            FragAB bv;
            int crow = ni * 16 + lm;
            const uint16_t* p = &vT[crow * 32 + lh * 8];
            bv.q[0] = *(const uint4*)p;
            bv.q[1] = *(const uint4*)(p + 16);
            acc2[ni] = __builtin_amdgcn_wmma_f32_16x16x32_bf16(
                           false, ap.v, false, bv.v, (short)0, acc2[ni], false, false);
        }
    }

    // ---- normalize and write a (bf16) in (B, C, T) layout
#pragma unroll
    for (int r = 0; r < 8; ++r) {
        float inv = 1.0f / l_run[r];
        int t = tblock + wave * 16 + lh * 8 + r;
#pragma unroll
        for (int ni = 0; ni < 4; ++ni) {
            int c = hd * CH_N + ni * 16 + lm;
            abf[((size_t)b * C_N + c) * T_N + t] = f2bf(acc2[ni][r] * inv);
        }
    }
}

// ---------------------------------------------------------------- launcher
extern "C" void kernel_launch(void* const* d_in, const int* in_sizes, int n_in,
                              void* d_out, int out_size, void* d_ws, size_t ws_size,
                              hipStream_t stream) {
    const float* x      = (const float*)d_in[0];
    const float* gn_w   = (const float*)d_in[1];
    const float* gn_b   = (const float*)d_in[2];
    const float* qkv_w  = (const float*)d_in[3];
    const float* qkv_b  = (const float*)d_in[4];
    const float* proj_w = (const float*)d_in[5];
    const float* proj_b = (const float*)d_in[6];
    float* out = (float*)d_out;

    // workspace carve-up (all 16B-aligned)
    char* ws = (char*)d_ws;
    float*    stats = (float*)ws;                                  // 64 floats
    uint16_t* wqkv  = (uint16_t*)(ws + 4096);                      // 1536*512
    uint16_t* wproj = wqkv  + (size_t)O3_N * C_N;                  // 512*512
    uint16_t* hbf   = wproj + (size_t)C_N * C_N;                   // B*C*T
    uint16_t* qkvb  = hbf   + (size_t)B_N * C_N * T_N;             // B*3C*T
    uint16_t* abf   = qkvb  + (size_t)B_N * O3_N * T_N;            // B*C*T

    cvt_bf_kernel<<<(O3_N * C_N) / 256, 256, 0, stream>>>(qkv_w, wqkv, O3_N * C_N);
    cvt_bf_kernel<<<(C_N * C_N) / 256, 256, 0, stream>>>(proj_w, wproj, C_N * C_N);
    gn_stats_kernel<<<B_N * G_N, 256, 0, stream>>>(x, stats);
    gn_apply_kernel<<<(B_N * C_N * T_N) / 256, 256, 0, stream>>>(x, gn_w, gn_b, stats, hbf);

    gemm_kernel<0><<<dim3(T_N / 128, O3_N / 128, B_N), 256, 0, stream>>>(
        wqkv, hbf, qkv_b, qkvb, nullptr, nullptr, C_N, O3_N);

    attn_kernel<<<dim3(T_N / 128, B_N * H_N), 256, 0, stream>>>(qkvb, abf);

    gemm_kernel<1><<<dim3(T_N / 128, C_N / 128, B_N), 256, 0, stream>>>(
        wproj, abf, proj_b, nullptr, x, out, C_N, C_N);
}
